// Decoder_55516747268299
// MI455X (gfx1250) — matrix operand order repaired and checked by reading, not compiled, not run on hardware
//
#include <hip/hip_runtime.h>
#include <math.h>

// ---------------- types for WMMA fragments ----------------
typedef __attribute__((ext_vector_type(16))) __bf16      v16bf;
typedef __attribute__((ext_vector_type(8)))  float       v8f;
typedef __attribute__((ext_vector_type(4)))  unsigned int v4u;

union FragAB { v16bf v; v4u q[2]; };
union U8     { v4u q; __bf16 b[8]; };

// Problem constants
#define Bb   128
#define Ss   512
#define Tt   128
#define Hh   512
#define He2  1024
#define Ed2  1024
#define Ee   512
#define KC   3072   // E + 2He + 2Ed + H  (rnn_in + h packed)
#define Gg   2048   // 4*H
#define KP   2048   // H + 2He + E (pre-output input)

// ---------------- elementwise prep kernels ----------------
__global__ void k_f2bf(const float* __restrict__ src, __bf16* __restrict__ dst, long n) {
    long i = (long)blockIdx.x * blockDim.x + threadIdx.x;
    if (i < n) dst[i] = (__bf16)src[i];
}

// src is R x C row-major fp32; dst is C x R row-major bf16 (transposed)
__global__ void k_transpose_bf16(const float* __restrict__ src, __bf16* __restrict__ dst,
                                 int R, int C) {
    long i = (long)blockIdx.x * blockDim.x + threadIdx.x;
    if (i >= (long)R * C) return;
    int r = (int)(i / C), c = (int)(i % C);
    dst[(size_t)c * R + r] = (__bf16)src[i];
}

// Combined recurrent weight: Wg_t[n][k], k<2560 -> W_ih[n][k], else W_hh[n][k-2560]
__global__ void k_build_wgt(const float* __restrict__ wih, const float* __restrict__ whh,
                            __bf16* __restrict__ dst) {
    long i = (long)blockIdx.x * blockDim.x + threadIdx.x;
    if (i >= (long)Gg * KC) return;
    int n = (int)(i / KC), k = (int)(i % KC);
    float v = (k < 2560) ? wih[(size_t)n * 2560 + k] : whh[(size_t)n * 512 + (k - 2560)];
    dst[i] = (__bf16)v;
}

// edit_flat -> static slot of rnn_in [1536:2560)
__global__ void k_edit_fill(const float* __restrict__ edit, __bf16* __restrict__ rnn_in) {
    int i = blockIdx.x * blockDim.x + threadIdx.x;  // < B*1024
    if (i >= Bb * Ed2) return;
    int b = i >> 10, k = i & 1023;
    rnn_in[(size_t)b * KC + 1536 + k] = (__bf16)edit[i];
}

// bridge inputs: [encoder_final | edit_final] and [encoder_cell | edit_cell] -> bf16
__global__ void k_bridge_in(const float* __restrict__ ef, const float* __restrict__ edf,
                            const float* __restrict__ ec, const float* __restrict__ edc,
                            __bf16* __restrict__ brh, __bf16* __restrict__ brc) {
    int i = blockIdx.x * blockDim.x + threadIdx.x;  // < B*2048
    if (i >= Bb * 2048) return;
    int b = i >> 11, k = i & 2047;
    float vh = (k < 1024) ? ef[(size_t)b * 1024 + k] : edf[(size_t)b * 1024 + (k - 1024)];
    float vc = (k < 1024) ? ec[(size_t)b * 1024 + k] : edc[(size_t)b * 1024 + (k - 1024)];
    brh[i] = (__bf16)vh;
    brc[i] = (__bf16)vc;
}

__global__ void k_bias(const float* __restrict__ bih, const float* __restrict__ bhh,
                       float* __restrict__ bg) {
    int i = blockIdx.x * blockDim.x + threadIdx.x;
    if (i < Gg) bg[i] = bih[i] + bhh[i];
}

// per-step: x_t -> rnn_in[0:512) and pre_in[0:512)
__global__ void k_xcopy(const __bf16* __restrict__ xbf, __bf16* __restrict__ rnn_in,
                        __bf16* __restrict__ pre_in, int t) {
    int i = blockIdx.x * blockDim.x + threadIdx.x;  // < B*E
    if (i >= Bb * Ee) return;
    int b = i >> 9, e = i & 511;
    __bf16 v = xbf[((size_t)b * Tt + t) * Ee + e];
    rnn_in[(size_t)b * KC + e] = v;
    pre_in[(size_t)b * KP + e] = v;
}

// ---------------- generic bf16 WMMA GEMM ----------------
// C[M,N] = act( A[M,K] @ Bw[N,K]^T + bias ), A/Bw bf16 row-major, fp32 accumulate.
// Block = 4 waves; block tile = 32(M) x 256(N); each wave owns a 32x64 slice.
// The shared 32-row A tile is staged to LDS per 32-K chunk with
// GLOBAL_LOAD_ASYNC_TO_LDS_B128 (ASYNCcnt path), then each wave reads its
// fragments with ds_load_b128. B fragments stream straight from L2, with a
// global_prefetch for the next K chunk. Requires M%32==0, N%256==0, K%32==0.
__global__ void k_wmma_gemm(const __bf16* __restrict__ A, long lda,
                            const __bf16* __restrict__ Bw, long ldb,
                            float* __restrict__ Cf, long ldcf,
                            __bf16* __restrict__ Cbf, long ldcbf,
                            const float* __restrict__ bias,
                            int M, int N, int K, int act, int tiles_n) {
    __shared__ __align__(16) char sA[32 * 64];   // 32 rows x 32 bf16 (one K chunk)
    (void)M;
    int wave = threadIdx.x >> 5;
    int lane = threadIdx.x & 31;
    int l = lane & 15, hi = lane >> 4;

    int tm = (int)blockIdx.x / tiles_n;
    int tn = (int)blockIdx.x % tiles_n;
    int m0 = tm * 32;
    int n0 = tn * 256 + wave * 64;   // this wave's 64-column slice

    v8f zero = {0.f, 0.f, 0.f, 0.f, 0.f, 0.f, 0.f, 0.f};
    v8f acc[2][4];
#pragma unroll
    for (int mt = 0; mt < 2; ++mt)
#pragma unroll
        for (int i = 0; i < 4; ++i) acc[mt][i] = zero;

    for (int k0 = 0; k0 < K; k0 += 32) {
        // ---- wave 0 stages the 32x32 A chunk into LDS asynchronously ----
        if (wave == 0) {
#pragma unroll
            for (int it = 0; it < 4; ++it) {
                int c    = it * 32 + lane;        // 128 x 16B chunks
                int row  = c >> 2;
                int coff = (c & 3) << 4;          // byte offset within row
                const char* gp =
                    (const char*)(A + (size_t)(m0 + row) * lda + k0) + coff;
                unsigned la = (unsigned)(unsigned long long)(sA + row * 64 + coff);
                asm volatile("global_load_async_to_lds_b128 %0, %1, off"
                             :: "v"(la), "v"(gp) : "memory");
            }
            asm volatile("s_wait_asynccnt 0x0" ::: "memory");
        }
        __syncthreads();

        // ---- A fragments from LDS (ISA 7.12.2 16-bit A 16x32 layout) ----
        FragAB a0, a1;
        {
            const v4u* p0 = (const v4u*)(sA + (size_t)l * 64 + hi * 16);
            a0.q[0] = p0[0];
            a0.q[1] = p0[2];   // +32 bytes: K offsets +16..+23
            const v4u* p1 = (const v4u*)(sA + (size_t)(16 + l) * 64 + hi * 16);
            a1.q[0] = p1[0];
            a1.q[1] = p1[2];
        }

        // prefetch next K chunk of this wave's B stream into caches
        if (k0 + 32 < K)
            __builtin_prefetch(Bw + (size_t)(n0 + l) * ldb + k0 + 32 + (hi << 4), 0, 3);

#pragma unroll
        for (int i = 0; i < 4; ++i) {
            // B fragment (32x16): lane l = column n0+i*16+l, K = 8*hi+0..7 and 16+8*hi+0..7
            const __bf16* brow = Bw + (size_t)(n0 + i * 16 + l) * ldb + k0 + (hi << 3);
            FragAB bf;
            bf.q[0] = *(const v4u*)(brow);
            bf.q[1] = *(const v4u*)(brow + 16);
            acc[0][i] = __builtin_amdgcn_wmma_f32_16x16x32_bf16(
                false, a0.v, false, bf.v, (short)0, acc[0][i], false, false);
            acc[1][i] = __builtin_amdgcn_wmma_f32_16x16x32_bf16(
                false, a1.v, false, bf.v, (short)0, acc[1][i], false, false);
        }
        __syncthreads();   // A chunk fully consumed before next overwrite
    }

    // C/D layout: VGPR v, lanes 0-15 -> M=+v, lanes 16-31 -> M=+v+8, N=+l
#pragma unroll
    for (int mt = 0; mt < 2; ++mt) {
#pragma unroll
        for (int i = 0; i < 4; ++i) {
#pragma unroll
            for (int v = 0; v < 8; ++v) {
                int row = m0 + mt * 16 + (hi << 3) + v;
                int col = n0 + i * 16 + l;
                float val = acc[mt][i][v];
                if (bias) val += bias[col];
                if (act)  val = tanhf(val);
                if (Cf)   Cf[(size_t)row * ldcf + col] = val;
                if (Cbf)  Cbf[(size_t)row * ldcbf + col] = (__bf16)val;
            }
        }
    }
}

// ---------------- attention (scores + softmax + context) ----------------
// One block per batch row; proj_key and encoder_hidden are bf16 (L2-resident).
__global__ void k_attn(const float* __restrict__ q, const __bf16* __restrict__ pk,
                       const __bf16* __restrict__ eh, const unsigned char* __restrict__ mask,
                       const float* __restrict__ wen,
                       __bf16* __restrict__ rnn_in, __bf16* __restrict__ pre_in) {
    __shared__ float qs[Hh];
    __shared__ float we[Hh];
    __shared__ float sc[Ss];
    __shared__ float red[256];
    int b = blockIdx.x, tid = threadIdx.x;

    for (int h = tid; h < Hh; h += 256) { qs[h] = q[(size_t)b * Hh + h]; we[h] = wen[h]; }
    __syncthreads();

    float myv[2];
#pragma unroll
    for (int i = 0; i < 2; ++i) {
        int s = tid + i * 256;
        const __bf16* row = pk + ((size_t)b * Ss + s) * Hh;
        __builtin_prefetch(row, 0, 3);
        float acc = 0.f;
        for (int h8 = 0; h8 < Hh; h8 += 8) {
            U8 u; u.q = *(const v4u*)(row + h8);
#pragma unroll
            for (int j = 0; j < 8; ++j)
                acc += tanhf(qs[h8 + j] + (float)u.b[j]) * we[h8 + j];
        }
        if (!mask[(size_t)b * Ss + s]) acc = -INFINITY;
        myv[i] = acc;
    }
    // block max
    red[tid] = fmaxf(myv[0], myv[1]);
    __syncthreads();
    for (int off = 128; off > 0; off >>= 1) {
        if (tid < off) red[tid] = fmaxf(red[tid], red[tid + off]);
        __syncthreads();
    }
    float smax = red[0];
    __syncthreads();
    float e0 = expf(myv[0] - smax), e1 = expf(myv[1] - smax);
    red[tid] = e0 + e1;
    __syncthreads();
    for (int off = 128; off > 0; off >>= 1) {
        if (tid < off) red[tid] += red[tid + off];
        __syncthreads();
    }
    float inv = 1.0f / red[0];
    __syncthreads();
    sc[tid] = e0 * inv;
    sc[tid + 256] = e1 * inv;
    __syncthreads();

    // context[b,d] = sum_s alphas[s] * eh[b,s,d]; coalesced over d
    for (int d = tid; d < He2; d += 256) {
        const __bf16* ehb = eh + (size_t)b * Ss * He2 + d;
        float acc = 0.f;
        for (int s = 0; s < Ss; ++s) acc += sc[s] * (float)ehb[(size_t)s * He2];
        rnn_in[(size_t)b * KC + 512 + d]  = (__bf16)acc;   // context slot of rnn_in
        pre_in[(size_t)b * KP + 1024 + d] = (__bf16)acc;   // context slot of pre_in
    }
}

// ---------------- LSTM cell elementwise ----------------
__global__ void k_cell(const float* __restrict__ gates, float* __restrict__ c,
                       __bf16* __restrict__ rnn_in, __bf16* __restrict__ pre_in,
                       float* __restrict__ states, float* __restrict__ hlast, int t) {
    int i = blockIdx.x * blockDim.x + threadIdx.x;  // < B*H
    if (i >= Bb * Hh) return;
    int b = i >> 9, j = i & 511;
    const float* g = gates + (size_t)b * Gg;
    float ig = g[j], fg = g[512 + j], gg = g[1024 + j], og = g[1536 + j];
    float si = 1.f / (1.f + expf(-ig));
    float sf = 1.f / (1.f + expf(-fg));
    float so = 1.f / (1.f + expf(-og));
    float cn = sf * c[i] + si * tanhf(gg);
    float hn = so * tanhf(cn);
    c[i] = cn;
    rnn_in[(size_t)b * KC + 2560 + j] = (__bf16)hn;   // h slot for next step's GEMMs
    pre_in[(size_t)b * KP + 512 + j]  = (__bf16)hn;   // h slot for pre-output
    states[((size_t)b * Tt + t) * Hh + j] = hn;       // decoder_states[b,t,:]
    hlast[i] = hn;                                     // overwritten; last step wins
}

// ---------------- host launcher ----------------
extern "C" void kernel_launch(void* const* d_in, const int* in_sizes, int n_in,
                              void* d_out, int out_size, void* d_ws, size_t ws_size,
                              hipStream_t stream) {
    (void)in_sizes; (void)n_in; (void)out_size; (void)ws_size;
    const float* trg     = (const float*)d_in[0];
    const float* ed_fin  = (const float*)d_in[1];
    const float* ed_cell = (const float*)d_in[2];
    const float* enc_hid = (const float*)d_in[3];
    const float* enc_fin = (const float*)d_in[4];
    const float* enc_cel = (const float*)d_in[5];
    const unsigned char* mask = (const unsigned char*)d_in[6];
    const float* W_key   = (const float*)d_in[7];
    const float* W_query = (const float*)d_in[8];
    const float* w_en    = (const float*)d_in[9];
    const float* W_bridge= (const float*)d_in[10];
    const float* b_bridge= (const float*)d_in[11];
    const float* W_ih    = (const float*)d_in[12];
    const float* W_hh    = (const float*)d_in[13];
    const float* b_ih    = (const float*)d_in[14];
    const float* b_hh    = (const float*)d_in[15];
    const float* W_pre   = (const float*)d_in[16];

    float* out_states = (float*)d_out;                               // (B,T,H)
    float* out_hlast  = out_states + (size_t)Bb * Tt * Hh;           // (1,B,H)
    float* out_pre    = out_hlast + (size_t)Bb * Hh;                 // (B,T,512)

    // workspace layout (bytes, 256-aligned)
    char* ws = (char*)d_ws;
    size_t off = 0;
    auto take = [&](size_t bytes) { char* p = ws + off; off += (bytes + 255) & ~(size_t)255; return p; };
    __bf16* eh_bf  = (__bf16*)take((size_t)Bb * Ss * He2 * 2);   // 134 MB
    __bf16* pk_bf  = (__bf16*)take((size_t)Bb * Ss * Hh * 2);    // 67 MB
    __bf16* x_bf   = (__bf16*)take((size_t)Bb * Tt * Ee * 2);    // 16.8 MB
    __bf16* Wk_t   = (__bf16*)take((size_t)Hh * He2 * 2);
    __bf16* Wq_t   = (__bf16*)take((size_t)Hh * Hh * 2);
    __bf16* Wb_t   = (__bf16*)take((size_t)Hh * 2048 * 2);
    __bf16* Wg_t   = (__bf16*)take((size_t)Gg * KC * 2);
    __bf16* Wp_bf  = (__bf16*)take((size_t)Hh * KP * 2);
    __bf16* rnn_in = (__bf16*)take((size_t)Bb * KC * 2);
    __bf16* pre_in = (__bf16*)take((size_t)Bb * KP * 2);
    __bf16* brh    = (__bf16*)take((size_t)Bb * 2048 * 2);
    __bf16* brc    = (__bf16*)take((size_t)Bb * 2048 * 2);
    float*  qbuf   = (float*)take((size_t)Bb * Hh * 4);
    float*  gates  = (float*)take((size_t)Bb * Gg * 4);
    float*  cbuf   = (float*)take((size_t)Bb * Hh * 4);
    float*  biasg  = (float*)take((size_t)Gg * 4);

    const int TH = 256;
    auto blk = [](long n, int th) { return (unsigned)((n + th - 1) / th); };

    // ---- one-time prep: bf16 conversions + weight transposes ----
    long n_eh = (long)Bb * Ss * He2;
    k_f2bf<<<blk(n_eh, TH), TH, 0, stream>>>(enc_hid, eh_bf, n_eh);
    long n_x = (long)Bb * Tt * Ee;
    k_f2bf<<<blk(n_x, TH), TH, 0, stream>>>(trg, x_bf, n_x);
    long n_wp = (long)Hh * KP;
    k_f2bf<<<blk(n_wp, TH), TH, 0, stream>>>(W_pre, Wp_bf, n_wp);       // already (N,K)
    k_transpose_bf16<<<blk((long)He2 * Hh, TH), TH, 0, stream>>>(W_key, Wk_t, He2, Hh);
    k_transpose_bf16<<<blk((long)Hh * Hh, TH), TH, 0, stream>>>(W_query, Wq_t, Hh, Hh);
    k_transpose_bf16<<<blk((long)2048 * Hh, TH), TH, 0, stream>>>(W_bridge, Wb_t, 2048, Hh);
    k_build_wgt<<<blk((long)Gg * KC, TH), TH, 0, stream>>>(W_ih, W_hh, Wg_t);
    k_edit_fill<<<blk((long)Bb * Ed2, TH), TH, 0, stream>>>(ed_fin, rnn_in);
    k_bridge_in<<<blk((long)Bb * 2048, TH), TH, 0, stream>>>(enc_fin, ed_fin, enc_cel, ed_cell,
                                                             brh, brc);
    k_bias<<<blk(Gg, TH), TH, 0, stream>>>(b_ih, b_hh, biasg);

    // bridge: h0 -> bf16 h-slot of rnn_in; c0 -> fp32 c buffer (tanh + bias fused)
    // M=128,N=512,K=2048 -> (128/32)*(512/256) = 8 blocks, tiles_n = 2
    k_wmma_gemm<<<8, 128, 0, stream>>>(brh, 2048, Wb_t, 2048,
                                       nullptr, 0, rnn_in + 2560, KC,
                                       b_bridge, Bb, Hh, 2048, 1, 2);
    k_wmma_gemm<<<8, 128, 0, stream>>>(brc, 2048, Wb_t, 2048,
                                       cbuf, Hh, nullptr, 0,
                                       b_bridge, Bb, Hh, 2048, 1, 2);
    // proj_key: (65536 x 1024) @ (1024 x 512) -> bf16 (L2-resident thereafter)
    // (65536/32)*(512/256) = 4096 blocks, tiles_n = 2
    k_wmma_gemm<<<4096, 128, 0, stream>>>(eh_bf, He2, Wk_t, He2,
                                          nullptr, 0, pk_bf, Hh,
                                          nullptr, Bb * Ss, Hh, He2, 0, 2);

    // ---- recurrent loop: 128 sequential steps ----
    for (int t = 0; t < Tt; ++t) {
        k_xcopy<<<blk((long)Bb * Ee, TH), TH, 0, stream>>>(x_bf, rnn_in, pre_in, t);
        // q = h @ W_query : A = h slot of rnn_in (lda=KC); 8 blocks
        k_wmma_gemm<<<8, 128, 0, stream>>>(rnn_in + 2560, KC, Wq_t, Hh,
                                           qbuf, Hh, nullptr, 0,
                                           nullptr, Bb, Hh, Hh, 0, 2);
        k_attn<<<Bb, 256, 0, stream>>>(qbuf, pk_bf, eh_bf, mask, w_en, rnn_in, pre_in);
        // gates = [x|ctx|edit|h] @ [W_ih|W_hh]^T + bias : (128/32)*(2048/256)=32 blocks
        k_wmma_gemm<<<32, 128, 0, stream>>>(rnn_in, KC, Wg_t, KC,
                                            gates, Gg, nullptr, 0,
                                            biasg, Bb, Gg, KC, 0, 8);
        k_cell<<<blk((long)Bb * Hh, TH), TH, 0, stream>>>(gates, cbuf, rnn_in, pre_in,
                                                          out_states, out_hlast, t);
        // pre = [x|h_new|ctx] @ W_pre^T -> out_pre[:, t, :] (row stride T*512); 8 blocks
        k_wmma_gemm<<<8, 128, 0, stream>>>(pre_in, KP, Wp_bf, KP,
                                           out_pre + (size_t)t * Hh, (long)Tt * Hh,
                                           nullptr, 0,
                                           nullptr, Bb, Hh, KP, 0, 2);
    }
}
